// AttentionPooling_vectorized_old_51041391345715
// MI455X (gfx1250) — matrix operand order repaired and checked by reading, not compiled, not run on hardware
//
#include <hip/hip_runtime.h>
#include <hip/hip_bf16.h>
#include <math.h>

// Problem constants (match the reference)
#define B_  2
#define T_  512
#define S_  2048
#define H_  768
#define L_  32
#define NH_ 12
#define HD_ 64
#define FF_ 3072   // 4*H

typedef __attribute__((ext_vector_type(16))) __bf16 v16bf;
typedef __attribute__((ext_vector_type(8)))  float  v8f;

__device__ __forceinline__ float wave_sum(float v) {
#pragma unroll
  for (int m = 16; m > 0; m >>= 1) v += __shfl_xor(v, m, 32);
  return v;
}

// Issue one 16B global->LDS async DMA (CDNA5 GLOBAL_LOAD_ASYNC_TO_LDS_B128,
// tracked by ASYNCcnt). ldsOff = LDS byte offset (generic ptr low 32 bits),
// gaddr = 64-bit global address.
__device__ __forceinline__ void async_copy_b128(unsigned ldsOff,
                                                unsigned long long gaddr) {
  asm volatile("global_load_async_to_lds_b128 %0, %1, off"
               :: "v"(ldsOff), "v"(gaddr) : "memory");
}
__device__ __forceinline__ void wait_async0() {
  asm volatile("s_wait_asynccnt 0x0" ::: "memory");
}

// ---------------------------------------------------------------------------
// Kernel A: q = Wq*dummy + bq ; u[n,h] = sum_d q[n*64+d]*Wk[n*64+d,h] ;
//           cK[n] = q_n . bk_n
// ---------------------------------------------------------------------------
__global__ __launch_bounds__(256) void precompute_kernel(
    const float* __restrict__ dq, const float* __restrict__ ipw,
    const float* __restrict__ ipb, float* __restrict__ u,
    float* __restrict__ cK) {
  __shared__ float sQ[H_];
  const int tid = threadIdx.x;
  for (int i = tid; i < H_; i += 256) {
    float acc = ipb[i];
    const float* wr = ipw + (size_t)i * H_;
    for (int h = 0; h < H_; ++h) acc += wr[h] * dq[h];
    sQ[i] = acc;
  }
  __syncthreads();
  for (int idx = tid; idx < NH_ * H_; idx += 256) {
    const int n = idx / H_, h = idx - n * H_;
    float acc = 0.f;
    const float* wk = ipw + (size_t)(H_ + n * HD_) * H_ + h;
#pragma unroll 4
    for (int d = 0; d < HD_; ++d) acc += sQ[n * HD_ + d] * wk[(size_t)d * H_];
    u[idx] = acc;
  }
  if (tid < NH_) {
    float acc = 0.f;
    for (int d = 0; d < HD_; ++d) acc += sQ[tid * HD_ + d] * ipb[H_ + tid * HD_ + d];
    cK[tid] = acc;
  }
}

// ---------------------------------------------------------------------------
// Kernel B: per-span fused attention pooling -> out_proj -> residual -> LN
// grid = B*S, block = 256 (8 wave32)
// ---------------------------------------------------------------------------
__global__ __launch_bounds__(256) void span_attn_kernel(
    const float* __restrict__ tr, const int* __restrict__ span_ids,
    const unsigned char* __restrict__ span_masks, const float* __restrict__ pe,
    const float* __restrict__ dq, const float* __restrict__ ipw,
    const float* __restrict__ ipb, const float* __restrict__ opw,
    const float* __restrict__ opb, const float* __restrict__ ng,
    const float* __restrict__ nb, const float* __restrict__ u,
    const float* __restrict__ cK,
    float* __restrict__ x1f, __bf16* __restrict__ x1bf) {
  __shared__ float sBuf[NH_ * H_];   // holds u, later pooled
  __shared__ float sAttn[NH_ * L_];
  __shared__ float sCtx[H_];
  __shared__ float sX1[H_];
  __shared__ float sRed[8];

  const int row = blockIdx.x;            // b*S + s
  const int bB  = row / S_;
  const int tid = threadIdx.x;
  const int lane = tid & 31, wave = tid >> 5;

  const int start = span_ids[2 * row];
  const int end   = span_ids[2 * row + 1];
  const int msk   = span_masks[row] ? 1 : 0;
  const int len   = (end - start) * msk;

  for (int i = tid; i < NH_ * H_; i += 256) sBuf[i] = u[i];
  __syncthreads();

  // ---- scores[n,l] = (u_n . g_l + cK[n]) / sqrt(64) -----------------------
  for (int l = wave; l < L_; l += 8) {
    int pos = start + l;
    pos = pos < 0 ? 0 : (pos >= T_ ? T_ - 1 : pos);
    const bool vld = (l < len);
    float part[NH_];
#pragma unroll
    for (int n = 0; n < NH_; ++n) part[n] = 0.f;
    if (vld) {
      const float* trr = tr + ((size_t)bB * T_ + pos) * H_;
      const float* per = pe + (size_t)pos * H_;
      for (int h = lane; h < H_; h += 32) {
        const float gv = trr[h] + per[h];
#pragma unroll
        for (int n = 0; n < NH_; ++n) part[n] += sBuf[n * H_ + h] * gv;
      }
    }
#pragma unroll
    for (int n = 0; n < NH_; ++n) {
      const float s = wave_sum(part[n]);
      if (lane == 0) sAttn[n * L_ + l] = (s + cK[n]) * 0.125f;
    }
  }
  __syncthreads();

  // ---- masked softmax per head (guard: fully-masked span keeps l=0) -------
  if (tid < NH_) {
    float mx = -3.0e38f;
    for (int l = 0; l < L_; ++l) {
      const bool vf = (l < len) || (l == 0 && len == 0);
      const float s = sAttn[tid * L_ + l];
      if (vf && s > mx) mx = s;
    }
    float sum = 0.f;
    for (int l = 0; l < L_; ++l) {
      const bool vf = (l < len) || (l == 0 && len == 0);
      const float e = vf ? __expf(sAttn[tid * L_ + l] - mx) : 0.f;
      sAttn[tid * L_ + l] = e;
      sum += e;
    }
    const float inv = 1.f / sum;
    for (int l = 0; l < L_; ++l) sAttn[tid * L_ + l] *= inv;
  }
  __syncthreads();

  // ---- pooled[n,h] = sum_l attn[n,l] * g[l,h] (g zeroed for invalid l) ----
  float acc[NH_][3];
#pragma unroll
  for (int n = 0; n < NH_; ++n) { acc[n][0] = acc[n][1] = acc[n][2] = 0.f; }
  const int lmax = len < L_ ? len : L_;
  for (int l = 0; l < lmax; ++l) {
    int pos = start + l;
    pos = pos < 0 ? 0 : (pos >= T_ ? T_ - 1 : pos);
    const float* trr = tr + ((size_t)bB * T_ + pos) * H_;
    const float* per = pe + (size_t)pos * H_;
    const float g0 = trr[tid] + per[tid];
    const float g1 = trr[tid + 256] + per[tid + 256];
    const float g2 = trr[tid + 512] + per[tid + 512];
#pragma unroll
    for (int n = 0; n < NH_; ++n) {
      const float a = sAttn[n * L_ + l];
      acc[n][0] += a * g0; acc[n][1] += a * g1; acc[n][2] += a * g2;
    }
  }
  __syncthreads();   // everyone done reading u in sBuf
#pragma unroll
  for (int n = 0; n < NH_; ++n) {
    sBuf[n * H_ + tid]       = acc[n][0];
    sBuf[n * H_ + tid + 256] = acc[n][1];
    sBuf[n * H_ + tid + 512] = acc[n][2];
  }
  __syncthreads();

  // ---- ctx[d] = Wv[d,:] . pooled[head(d),:] + bv[d] -----------------------
  for (int d = wave; d < H_; d += 8) {
    const int n = d >> 6;
    const float* wv = ipw + (size_t)(2 * H_ + d) * H_;
    float a = 0.f;
    for (int h = lane; h < H_; h += 32) a += wv[h] * sBuf[n * H_ + h];
    a = wave_sum(a);
    if (lane == 0) sCtx[d] = a + ipb[2 * H_ + d];
  }
  __syncthreads();

  // ---- attn_out + residual(dummy_query) -----------------------------------
  for (int o = wave; o < H_; o += 8) {
    const float* wr = opw + (size_t)o * H_;
    float a = 0.f;
    for (int d = lane; d < H_; d += 32) a += wr[d] * sCtx[d];
    a = wave_sum(a);
    if (lane == 0) sX1[o] = a + opb[o] + dq[o];
  }
  __syncthreads();

  // ---- LayerNorm ----------------------------------------------------------
  float lsum = 0.f;
  for (int o = tid; o < H_; o += 256) lsum += sX1[o];
  lsum = wave_sum(lsum);
  if (lane == 0) sRed[wave] = lsum;
  __syncthreads();
  float mu = 0.f;
#pragma unroll
  for (int w = 0; w < 8; ++w) mu += sRed[w];
  mu *= (1.f / H_);
  __syncthreads();
  float lvar = 0.f;
  for (int o = tid; o < H_; o += 256) { const float d = sX1[o] - mu; lvar += d * d; }
  lvar = wave_sum(lvar);
  if (lane == 0) sRed[wave] = lvar;
  __syncthreads();
  float var = 0.f;
#pragma unroll
  for (int w = 0; w < 8; ++w) var += sRed[w];
  var *= (1.f / H_);
  const float rs = rsqrtf(var + 1e-5f);
  for (int o = tid; o < H_; o += 256) {
    const float v = (sX1[o] - mu) * rs * ng[o] + nb[o];
    x1f[(size_t)row * H_ + o]  = v;
    x1bf[(size_t)row * H_ + o] = (__bf16)v;
  }
}

// ---------------------------------------------------------------------------
// Elementwise f32 -> bf16 conversion (weights)
// ---------------------------------------------------------------------------
__global__ __launch_bounds__(256) void f32_to_bf16_kernel(
    const float* __restrict__ in, __bf16* __restrict__ out, int n) {
  const int i = blockIdx.x * 256 + threadIdx.x;
  if (i < n) out[i] = (__bf16)in[i];
}

// ---------------------------------------------------------------------------
// Kernel D: C = act(A @ W^T + bias). A: MxK bf16 row-major, W: NxK bf16
// row-major (WMMA B matrix element (k,n) = W[n,k]).
// Block tile 64x128, K-step 32, 8 waves; wave = 16x64 strip = 4 WMMAs/K-step.
// Tiles are DMA'd with global_load_async_to_lds_b128 into double-buffered
// LDS: issue tile k+1 right after the barrier, compute tile k -> one barrier
// per K-step and a full compute phase of latency hiding (ASYNCcnt drain).
// ---------------------------------------------------------------------------
template <bool RELU, bool OUT_BF16>
__global__ __launch_bounds__(256) void wmma_gemm_kernel(
    const __bf16* __restrict__ A, const __bf16* __restrict__ W,
    const float* __restrict__ bias,
    float* __restrict__ Cf, __bf16* __restrict__ Cb,
    int M, int N, int K) {
  __shared__ __align__(16) __bf16 sA[2][64 * 32];
  __shared__ __align__(16) __bf16 sB[2][128 * 32];
  const int tid  = threadIdx.x;
  const int lane = tid & 31, wave = tid >> 5;
  const int wm = wave >> 1, wn = wave & 1;
  const int half = lane >> 4, l16 = lane & 15;
  const int m_blk = blockIdx.y * 64;
  const int n_blk = blockIdx.x * 128;

  v8f acc[4];
#pragma unroll
  for (int j = 0; j < 4; ++j)
#pragma unroll
    for (int r = 0; r < 8; ++r) acc[j][r] = 0.f;

  // Per-thread DMA chunk assignment: A tile = 256 x 16B, B tile = 512 x 16B.
  const int arow = tid >> 2;            // 0..63
  const int akc  = (tid & 3) * 8;       // 0,8,16,24
  const int brow0 = tid >> 2;           // chunk c=tid
  const int bkc0  = (tid & 3) * 8;
  const int brow1 = (tid + 256) >> 2;   // chunk c=tid+256
  const int bkc1  = ((tid + 256) & 3) * 8;

  const __bf16* aSrc  = A + (size_t)(m_blk + arow)  * K + akc;
  const __bf16* bSrc0 = W + (size_t)(n_blk + brow0) * K + bkc0;
  const __bf16* bSrc1 = W + (size_t)(n_blk + brow1) * K + bkc1;
  const unsigned ldsA  = (unsigned)(uintptr_t)(&sA[0][arow * 32 + akc]);
  const unsigned ldsB0 = (unsigned)(uintptr_t)(&sB[0][brow0 * 32 + bkc0]);
  const unsigned ldsB1 = (unsigned)(uintptr_t)(&sB[0][brow1 * 32 + bkc1]);
  const unsigned bufStrideA = (unsigned)(64 * 32 * sizeof(__bf16));   // 4KB
  const unsigned bufStrideB = (unsigned)(128 * 32 * sizeof(__bf16));  // 8KB

  auto issue_tiles = [&](int buf, int k0) {
    async_copy_b128(ldsA  + (unsigned)buf * bufStrideA,
                    (unsigned long long)(uintptr_t)(aSrc + k0));
    async_copy_b128(ldsB0 + (unsigned)buf * bufStrideB,
                    (unsigned long long)(uintptr_t)(bSrc0 + k0));
    async_copy_b128(ldsB1 + (unsigned)buf * bufStrideB,
                    (unsigned long long)(uintptr_t)(bSrc1 + k0));
  };

  issue_tiles(0, 0);
  int buf = 0;
  for (int k0 = 0; k0 < K; k0 += 32) {
    wait_async0();       // this wave's DMAs into sA/sB[buf] have landed
    __syncthreads();     // ... and everyone else's too
    if (k0 + 32 < K) issue_tiles(buf ^ 1, k0 + 32);  // prefetch next tile

    // A fragment (16x32): lane<16 row m, elems = K{0..7,16..23};
    //                     lane>=16 same row, elems = K{8..15,24..31}
    v16bf af;
    {
      const __bf16* ap = &sA[buf][(wm * 16 + l16) * 32 + half * 8];
#pragma unroll
      for (int i = 0; i < 8; ++i) af[i] = ap[i];
#pragma unroll
      for (int i = 0; i < 8; ++i) af[8 + i] = ap[16 + i];
    }
#pragma unroll
    for (int j = 0; j < 4; ++j) {
      // B fragment (32x16): lane<16 col n, K{0..7,16..23}; lane>=16 col n, K{8..15,24..31}
      v16bf bfv;
      const __bf16* bp = &sB[buf][(wn * 64 + j * 16 + l16) * 32 + half * 8];
#pragma unroll
      for (int i = 0; i < 16; ++i) bfv[i] = bp[(i & 8) * 2 + (i & 7)];
      acc[j] = __builtin_amdgcn_wmma_f32_16x16x32_bf16(
          false, af, false, bfv, (short)0, acc[j], false, false);
    }
    buf ^= 1;
    // No trailing barrier needed: the next iteration's wait+barrier happens
    // before anyone overwrites the buffer we just read.
  }

  // Epilogue: C layout — lane gives N, VGPR r gives M=r (lanes<16) / 8+r
#pragma unroll
  for (int j = 0; j < 4; ++j) {
    const int col = n_blk + wn * 64 + j * 16 + l16;
    const float bsv = bias[col];
#pragma unroll
    for (int r = 0; r < 8; ++r) {
      const int rowi = m_blk + wm * 16 + half * 8 + r;
      float v = acc[j][r] + bsv;
      if (RELU) v = v > 0.f ? v : 0.f;
      if (OUT_BF16) Cb[(size_t)rowi * N + col] = (__bf16)v;
      else          Cf[(size_t)rowi * N + col] = v;
    }
  }
}

// ---------------------------------------------------------------------------
// Kernel E: out = LayerNorm(y2 + x1) * mask
// ---------------------------------------------------------------------------
__global__ __launch_bounds__(256) void final_ln_kernel(
    const float* __restrict__ y2, const float* __restrict__ x1f,
    const unsigned char* __restrict__ span_masks,
    const float* __restrict__ ng, const float* __restrict__ nb,
    float* __restrict__ out) {
  __shared__ float sZ[H_];
  __shared__ float sRed[8];
  const int row = blockIdx.x;
  const int tid = threadIdx.x, lane = tid & 31, wave = tid >> 5;

  float lsum = 0.f;
  for (int o = tid; o < H_; o += 256) {
    const float z = y2[(size_t)row * H_ + o] + x1f[(size_t)row * H_ + o];
    sZ[o] = z; lsum += z;
  }
  lsum = wave_sum(lsum);
  if (lane == 0) sRed[wave] = lsum;
  __syncthreads();
  float mu = 0.f;
#pragma unroll
  for (int w = 0; w < 8; ++w) mu += sRed[w];
  mu *= (1.f / H_);
  __syncthreads();
  float lvar = 0.f;
  for (int o = tid; o < H_; o += 256) { const float d = sZ[o] - mu; lvar += d * d; }
  lvar = wave_sum(lvar);
  if (lane == 0) sRed[wave] = lvar;
  __syncthreads();
  float var = 0.f;
#pragma unroll
  for (int w = 0; w < 8; ++w) var += sRed[w];
  var *= (1.f / H_);
  const float rs = rsqrtf(var + 1e-5f);
  const float mk = span_masks[row] ? 1.f : 0.f;
  for (int o = tid; o < H_; o += 256)
    out[(size_t)row * H_ + o] = ((sZ[o] - mu) * rs * ng[o] + nb[o]) * mk;
}

// ---------------------------------------------------------------------------
extern "C" void kernel_launch(void* const* d_in, const int* in_sizes, int n_in,
                              void* d_out, int out_size, void* d_ws, size_t ws_size,
                              hipStream_t stream) {
  (void)in_sizes; (void)n_in; (void)out_size; (void)ws_size;
  const float*         token_reps = (const float*)d_in[0];
  const int*           span_ids   = (const int*)d_in[1];
  const unsigned char* span_masks = (const unsigned char*)d_in[2]; // bool array
  /* d_in[3] = pooling (unused, ==0) */
  const float* pe  = (const float*)d_in[4];
  const float* dq  = (const float*)d_in[5];
  const float* ipw = (const float*)d_in[6];
  const float* ipb = (const float*)d_in[7];
  const float* opw = (const float*)d_in[8];
  const float* opb = (const float*)d_in[9];
  const float* ng  = (const float*)d_in[10];
  const float* nb  = (const float*)d_in[11];
  const float* w1  = (const float*)d_in[12];
  const float* b1  = (const float*)d_in[13];
  const float* w2  = (const float*)d_in[14];
  const float* b2  = (const float*)d_in[15];
  float* out = (float*)d_out;

  // Workspace carve-up (256B aligned slices)
  char* ws = (char*)d_ws;
  size_t off = 0;
  auto alloc = [&](size_t bytes) -> void* {
    void* p = ws + off;
    off = (off + bytes + 255) & ~(size_t)255;
    return p;
  };
  float*  u   = (float*)alloc((size_t)NH_ * H_ * 4);
  float*  cK  = (float*)alloc((size_t)NH_ * 4);
  float*  x1f = (float*)alloc((size_t)B_ * S_ * H_ * 4);
  __bf16* x1b = (__bf16*)alloc((size_t)B_ * S_ * H_ * 2);
  __bf16* w1b = (__bf16*)alloc((size_t)FF_ * H_ * 2);
  __bf16* w2b = (__bf16*)alloc((size_t)H_ * FF_ * 2);
  __bf16* y1b = (__bf16*)alloc((size_t)B_ * S_ * FF_ * 2);
  float*  y2f = (float*)alloc((size_t)B_ * S_ * H_ * 4);

  precompute_kernel<<<1, 256, 0, stream>>>(dq, ipw, ipb, u, cK);

  span_attn_kernel<<<B_ * S_, 256, 0, stream>>>(
      token_reps, span_ids, span_masks, pe, dq, ipw, ipb, opw, opb, ng, nb,
      u, cK, x1f, x1b);

  const int nw = FF_ * H_;  // 2359296 for both weight matrices
  f32_to_bf16_kernel<<<(nw + 255) / 256, 256, 0, stream>>>(w1, w1b, nw);
  f32_to_bf16_kernel<<<(nw + 255) / 256, 256, 0, stream>>>(w2, w2b, nw);

  dim3 g1(FF_ / 128, (B_ * S_) / 64);   // (24, 64)
  wmma_gemm_kernel<true, true><<<g1, 256, 0, stream>>>(
      x1b, w1b, b1, nullptr, y1b, B_ * S_, FF_, H_);

  dim3 g2(H_ / 128, (B_ * S_) / 64);    // (6, 64)
  wmma_gemm_kernel<false, false><<<g2, 256, 0, stream>>>(
      y1b, w2b, b2, y2f, nullptr, B_ * S_, H_, FF_);

  final_ln_kernel<<<B_ * S_, 256, 0, stream>>>(y2f, x1f, span_masks, ng, nb, out);
}
